// Efficient_SKAN_grouped_16655883173894
// MI455X (gfx1250) — matrix operand order repaired and checked by reading, compiled, not run on hardware
//
#include <hip/hip_runtime.h>

typedef __attribute__((ext_vector_type(16))) _Float16 v16h;
typedef __attribute__((ext_vector_type(8)))  float    v8f;

#define IN_F  128
#define OUT_F 128
#define NUM_F 8
#define SEGS  9            // 1 base (silu) + 8 sin segments
#define KTILES 4           // 128 / 32 K-steps per segment
#define NTILES 8           // 128 / 16 output tiles
#define ROWS_PER_WAVE 16
#define WAVES_PER_BLOCK 4
#define ROWS_PER_BLOCK (ROWS_PER_WAVE * WAVES_PER_BLOCK)
#define LDS_STRIDE 132     // 128 + 4 pad: stride-128 column reads would be 16-way bank conflicts
#define WPK_HALFS (SEGS * KTILES * NTILES * 512)   // 147456 f16 = 288 KB

// ---------------------------------------------------------------------------
// Prep 1: pack folded weights into WMMA B-fragment layout.
// Block (s,kt,nt) holds a 32(K)x16(N) f16 tile as [lane][h] with
//   K = kt*32 + (lane>=16 ? 16 : 0) + h,   N = nt*16 + (lane&15)
// Weight folding:  seg 0      -> base_weight[o][i]              (no scale)
//                  seg s=f+1  -> conv_w[f][o][i] * coef[i/16][f] * scale_sp[o]
// ---------------------------------------------------------------------------
__global__ void skan_prep_weights(const float* __restrict__ base_w,
                                  const float* __restrict__ scale_sp,
                                  const float* __restrict__ coef,
                                  const float* __restrict__ conv_w,
                                  _Float16* __restrict__ wpk)
{
    int idx = blockIdx.x * blockDim.x + threadIdx.x;
    if (idx >= WPK_HALFS) return;
    int h  = idx & 15;
    int l  = (idx >> 4) & 31;
    int bi = idx >> 9;                 // (s*4 + kt)*8 + nt
    int nt = bi & 7;
    int kb = bi >> 3;
    int kt = kb & 3;
    int s  = kb >> 2;
    int o   = nt * 16 + (l & 15);
    int kin = kt * 32 + ((l >= 16) ? 16 : 0) + h;   // feature index within segment
    float v;
    if (s == 0) {
        v = base_w[o * IN_F + kin];
    } else {
        int f = s - 1;
        int g = kin >> 4;                            // group of 16 channels
        v = conv_w[((size_t)f * OUT_F + o) * IN_F + kin] * coef[g * NUM_F + f] * scale_sp[o];
    }
    wpk[idx] = (_Float16)v;
}

// Prep 2: fused spline bias  bias[o] = scale_sp[o] * sum_f conv_b[f][o]
__global__ void skan_prep_bias(const float* __restrict__ scale_sp,
                               const float* __restrict__ conv_b,
                               float* __restrict__ bias)
{
    int o = threadIdx.x;               // 128 threads
    float s = 0.f;
#pragma unroll
    for (int f = 0; f < NUM_F; ++f) s += conv_b[f * OUT_F + o];
    bias[o] = s * scale_sp[o];
}

// ---------------------------------------------------------------------------
// Main fused kernel: per wave, 16 rows x 128 outputs, K = 1152 via WMMA f16.
// ---------------------------------------------------------------------------
__global__ __launch_bounds__(128) void skan_main(const float* __restrict__ x,
                                                 const float* __restrict__ grid,
                                                 const _Float16* __restrict__ wpk,
                                                 const float* __restrict__ bias,
                                                 float* __restrict__ out)
{
    __shared__ float xs[WAVES_PER_BLOCK][ROWS_PER_WAVE][LDS_STRIDE];
    const int tid  = threadIdx.x;
    const int wave = tid >> 5;
    const int lane = tid & 31;
    const int rowBase = (blockIdx.x * WAVES_PER_BLOCK + wave) * ROWS_PER_WAVE;

    // ---- stage this wave's 16x128 f32 x-tile into LDS (coalesced b128) ----
    {
        const float4* xg = (const float4*)(x + (size_t)rowBase * IN_F);
#pragma unroll
        for (int t = 0; t < 16; ++t) {
            int linear = t * 32 + lane;        // 512 float4s per tile
            int row = linear >> 5;             // 32 float4 per row
            int c4  = linear & 31;
            float4 v = xg[linear];
            *(float4*)&xs[wave][row][c4 * 4] = v;
        }
    }
    __syncthreads();

    v8f acc[NTILES];
#pragma unroll
    for (int nt = 0; nt < NTILES; ++nt) acc[nt] = (v8f){};

    const int   m    = lane & 15;              // A-matrix row / C column within tile
    const int   offA = (lane >= 16) ? 16 : 0;  // A-fragment K offset for hi lanes
    const float* xrow = &xs[wave][m][0];

    // ---- segment 0: SiLU ----
#pragma unroll
    for (int kt = 0; kt < KTILES; ++kt) {
        const float4 q0 = *(const float4*)(xrow + kt * 32 + offA);
        const float4 q1 = *(const float4*)(xrow + kt * 32 + offA + 4);
        const float4 q2 = *(const float4*)(xrow + kt * 32 + offA + 8);
        const float4 q3 = *(const float4*)(xrow + kt * 32 + offA + 12);
        const float xa[16] = {q0.x,q0.y,q0.z,q0.w, q1.x,q1.y,q1.z,q1.w,
                              q2.x,q2.y,q2.z,q2.w, q3.x,q3.y,q3.z,q3.w};
        v16h a;
#pragma unroll
        for (int h = 0; h < 16; ++h) {
            float xv = xa[h];
            float sv = xv * __builtin_amdgcn_rcpf(1.0f + __expf(-xv));   // silu
            a[h] = (_Float16)sv;
        }
        const _Float16* wseg = wpk + (size_t)(0 * KTILES + kt) * NTILES * 512;
#pragma unroll
        for (int nt = 0; nt < NTILES; ++nt) {
            v16h b = *(const v16h*)(wseg + nt * 512 + lane * 16);
            acc[nt] = __builtin_amdgcn_wmma_f32_16x16x32_f16(
                false, a, false, b, (short)0, acc[nt], false, false);
        }
    }

    // ---- segments 1..8: sin(grid[f] * x), coef/scale folded into weights ----
    for (int s = 1; s < SEGS; ++s) {
        const float g = grid[s - 1];
#pragma unroll
        for (int kt = 0; kt < KTILES; ++kt) {
            const float4 q0 = *(const float4*)(xrow + kt * 32 + offA);
            const float4 q1 = *(const float4*)(xrow + kt * 32 + offA + 4);
            const float4 q2 = *(const float4*)(xrow + kt * 32 + offA + 8);
            const float4 q3 = *(const float4*)(xrow + kt * 32 + offA + 12);
            const float xa[16] = {q0.x,q0.y,q0.z,q0.w, q1.x,q1.y,q1.z,q1.w,
                                  q2.x,q2.y,q2.z,q2.w, q3.x,q3.y,q3.z,q3.w};
            v16h a;
#pragma unroll
            for (int h = 0; h < 16; ++h)
                a[h] = (_Float16)__sinf(g * xa[h]);
            const _Float16* wseg = wpk + (size_t)(s * KTILES + kt) * NTILES * 512;
#pragma unroll
            for (int nt = 0; nt < NTILES; ++nt) {
                v16h b = *(const v16h*)(wseg + nt * 512 + lane * 16);
                acc[nt] = __builtin_amdgcn_wmma_f32_16x16x32_f16(
                    false, a, false, b, (short)0, acc[nt], false, false);
            }
        }
    }

    // ---- epilogue: C layout -> global, + fused spline bias ----
    const int offM = (lane >= 16) ? 8 : 0;     // C rows for hi lanes
#pragma unroll
    for (int nt = 0; nt < NTILES; ++nt) {
        int o = nt * 16 + m;
        float bo = bias[o];
#pragma unroll
        for (int r = 0; r < 8; ++r) {
            out[(size_t)(rowBase + r + offM) * OUT_F + o] = acc[nt][r] + bo;
        }
    }
}

extern "C" void kernel_launch(void* const* d_in, const int* in_sizes, int n_in,
                              void* d_out, int out_size, void* d_ws, size_t ws_size,
                              hipStream_t stream) {
    const float* x        = (const float*)d_in[0];
    const float* grid     = (const float*)d_in[1];
    const float* base_w   = (const float*)d_in[2];
    const float* scale_sp = (const float*)d_in[3];
    const float* coef     = (const float*)d_in[4];
    const float* conv_w   = (const float*)d_in[5];
    const float* conv_b   = (const float*)d_in[6];
    float* out = (float*)d_out;

    _Float16* wpk = (_Float16*)d_ws;
    float* bias   = (float*)((char*)d_ws + (size_t)WPK_HALFS * sizeof(_Float16));

    skan_prep_weights<<<(WPK_HALFS + 255) / 256, 256, 0, stream>>>(base_w, scale_sp, coef, conv_w, wpk);
    skan_prep_bias<<<1, 128, 0, stream>>>(scale_sp, conv_b, bias);

    int rows   = in_sizes[0] / IN_F;           // 65536
    int blocks = rows / ROWS_PER_BLOCK;        // 1024
    skan_main<<<blocks, 128, 0, stream>>>(x, grid, wpk, bias, out);
}
